// GCN_1065151889943
// MI455X (gfx1250) — compile-verified, matrix operand order checked by reading, not run on hardware
//
#include <hip/hip_runtime.h>
#include <hip/hip_bf16.h>

typedef __attribute__((ext_vector_type(16))) __bf16 v16bf;
typedef __attribute__((ext_vector_type(4)))  __bf16 v4bf;
typedef __attribute__((ext_vector_type(8)))  float  v8f;

#define NN 100000

// ---------------- utility kernels ----------------

__global__ void zero_kernel(float* __restrict__ p, long n) {
  long i = (long)blockIdx.x * blockDim.x + threadIdx.x;
  long stride = (long)gridDim.x * blockDim.x;
  for (; i < n; i += stride) p[i] = 0.0f;
}

__global__ void deg_kernel(const int* __restrict__ src, const int* __restrict__ dst,
                           float* __restrict__ degO, float* __restrict__ degI, int E) {
  int e = blockIdx.x * blockDim.x + threadIdx.x;
  if (e < E) {
    atomicAdd(&degO[src[e]], 1.0f);
    atomicAdd(&degI[dst[e]], 1.0f);
  }
}

__global__ void norm_kernel(float* __restrict__ d, int n) {
  int i = blockIdx.x * blockDim.x + threadIdx.x;
  if (i < n) {
    float v = d[i];
    d[i] = (v > 0.0f) ? rsqrtf(v) : 0.0f;
  }
}

// ---------------- W pre-pack: f32 row-major -> bf16 WMMA-B fragment order ---
// Wp[((kt*CT + ctile)*32 + lane)*16 + j], lane = h*16 + n, where for K index k:
// kt=k/32, kk=k%32, g=kk>>3, h=g&1, j=(kk&7)+8*(g>>1).

__global__ void pack_w_kernel(const float* __restrict__ W, __bf16* __restrict__ Wp,
                              int K, int M) {
  int i = blockIdx.x * blockDim.x + threadIdx.x;
  if (i >= K * M) return;
  int k = i / M, col = i % M;
  int kt = k >> 5, kk = k & 31;
  int g = kk >> 3, h = g & 1, j = (kk & 7) + ((g >> 1) << 3);
  int ctile = col >> 4, n = col & 15;
  int CT = M >> 4;
  long dst = (((long)kt * CT + ctile) * 32 + (h * 16 + n)) * 16 + j;
  Wp[dst] = (__bf16)W[i];
}

// ---------------- WMMA GEMM:  H = (X @ W) * norm_src[row] ----------------
// 256 threads = 8 waves; 128 rows per block; wave w owns row-tile w, sweeps all
// column tiles. Double-buffered fragment-layout LDS staging of X; all B
// fragments batch-loaded into registers so WMMAs issue back-to-back.

template <int MOUT>
__global__ __launch_bounds__(256)
void gemm_norm_kernel(const float* __restrict__ X, const __bf16* __restrict__ Wp,
                      const float* __restrict__ nrm, float* __restrict__ H,
                      int Nn, int K) {
  constexpr int CT = MOUT / 16;
  __shared__ __bf16 Xs[2][8 * 32 * 16];   // 2 x 8KB, fragment layout

  const int tid  = threadIdx.x;
  const int wave = tid >> 5;
  const int lane = tid & 31;
  const int m    = lane & 15;
  const int h    = lane >> 4;
  const int rbase = blockIdx.x * 128;

  // per-thread staging coordinates (quad q = u*256 + tid)
  int rloc[4], kqv[4], dsoff[4];
#pragma unroll
  for (int u = 0; u < 4; ++u) {
    int q = u * 256 + tid;
    int r = q >> 3, kq = q & 7;
    int g = kq >> 1, hh = g & 1;
    int j0 = ((kq & 1) << 2) + ((g >> 1) << 3);
    int ln = (hh << 4) + (r & 15);
    int tile = r >> 4;
    rloc[u] = r; kqv[u] = kq;
    dsoff[u] = (((tile << 5) + ln) << 4) + j0;
  }

  v8f acc[CT];
#pragma unroll
  for (int c = 0; c < CT; ++c) acc[c] = (v8f){0, 0, 0, 0, 0, 0, 0, 0};

  const int ksteps = K >> 5;

  // prologue: stage k-step 0
  float4 stg[4];
#pragma unroll
  for (int u = 0; u < 4; ++u) {
    int row = rbase + rloc[u];
    stg[u] = (row < Nn) ? *(const float4*)&X[(long)row * K + (kqv[u] << 2)]
                        : make_float4(0.f, 0.f, 0.f, 0.f);
  }
#pragma unroll
  for (int u = 0; u < 4; ++u) {
    v4bf p;
    p[0] = (__bf16)stg[u].x; p[1] = (__bf16)stg[u].y;
    p[2] = (__bf16)stg[u].z; p[3] = (__bf16)stg[u].w;
    *(v4bf*)&Xs[0][dsoff[u]] = p;
  }
  __syncthreads();

  for (int kt = 0; kt < ksteps; ++kt) {
    const int buf = kt & 1;
    const bool more = (kt + 1) < ksteps;

    // prefetch next X tile into registers (overlaps with WMMA chain below)
    float4 nstg[4];
    if (more) {
#pragma unroll
      for (int u = 0; u < 4; ++u) {
        int row = rbase + rloc[u];
        nstg[u] = (row < Nn)
            ? *(const float4*)&X[(long)row * K + ((kt + 1) << 5) + (kqv[u] << 2)]
            : make_float4(0.f, 0.f, 0.f, 0.f);
      }
    }

    // batch-load all B fragments (distinct registers -> one wait, no serialization)
    const __bf16* wp = Wp + (long)kt * CT * 512;
    v16bf b[CT];
#pragma unroll
    for (int c = 0; c < CT; ++c)
      b[c] = *(const v16bf*)&wp[((c << 5) + lane) << 4];

    // A fragment: one contiguous 32B LDS read
    v16bf a = *(const v16bf*)&Xs[buf][((wave << 5) + lane) << 4];

    // back-to-back WMMAs on independent accumulators
#pragma unroll
    for (int c = 0; c < CT; ++c)
      acc[c] = __builtin_amdgcn_wmma_f32_16x16x32_bf16(
          false, a, false, b[c], (short)0, acc[c], false, false);

    // commit next tile to the other LDS buffer
    if (more) {
#pragma unroll
      for (int u = 0; u < 4; ++u) {
        v4bf p;
        p[0] = (__bf16)nstg[u].x; p[1] = (__bf16)nstg[u].y;
        p[2] = (__bf16)nstg[u].z; p[3] = (__bf16)nstg[u].w;
        *(v4bf*)&Xs[buf ^ 1][dsoff[u]] = p;
      }
    }
    __syncthreads();
  }

  // epilogue: D element r -> row (r + 8h), col (c*16 + m); fuse norm_src scale
#pragma unroll
  for (int r = 0; r < 8; ++r) {
    int row = rbase + wave * 16 + r + 8 * h;
    if (row < Nn) {
      float s = nrm[row];
#pragma unroll
      for (int c = 0; c < CT; ++c)
        H[(long)row * MOUT + (c * 16 + m)] = acc[c][r] * s;
    }
  }
}

// ---------------- edge scatter-add:  A[dst] += H[src]  ----------------

template <int TPE>
__global__ void agg_kernel(const float* __restrict__ H, float* __restrict__ A,
                           const int* __restrict__ src, const int* __restrict__ dst,
                           int E) {
  long tid = (long)blockIdx.x * blockDim.x + threadIdx.x;
  long e = tid / TPE;
  int  c = (int)(tid % TPE) * 4;
  if (e < E) {
    int s = src[e], d = dst[e];
    const float4 v = *(const float4*)&H[(long)s * (TPE * 4) + c];
    float* o = &A[(long)d * (TPE * 4) + c];
    atomicAdd(o + 0, v.x);
    atomicAdd(o + 1, v.y);
    atomicAdd(o + 2, v.z);
    atomicAdd(o + 3, v.w);
  }
}

// ---------------- per-node normalize + bias (+ReLU), in place ----------------

template <int F, bool RELU>
__global__ void finish_kernel(float* __restrict__ A, const float* __restrict__ nrmd,
                              const float* __restrict__ bias, int Nn) {
  long i = (long)blockIdx.x * blockDim.x + threadIdx.x;
  long n = (long)Nn * F;
  if (i < n) {
    int row = (int)(i / F);
    int c   = (int)(i % F);
    float v = A[i] * nrmd[row] + bias[c];
    if (RELU) v = fmaxf(v, 0.0f);
    A[i] = v;
  }
}

// ---------------- launcher ----------------

extern "C" void kernel_launch(void* const* d_in, const int* in_sizes, int n_in,
                              void* d_out, int out_size, void* d_ws, size_t ws_size,
                              hipStream_t stream) {
  const float* features = (const float*)d_in[0];
  const int*   src      = (const int*)d_in[1];
  const int*   dst      = (const int*)d_in[2];
  const float* W1       = (const float*)d_in[3];
  const float* b1       = (const float*)d_in[4];
  const float* W2       = (const float*)d_in[5];
  const float* b2       = (const float*)d_in[6];
  const float* W3       = (const float*)d_in[7];
  const float* b3       = (const float*)d_in[8];
  float* out = (float*)d_out;

  const int Nn = NN;
  const int E  = in_sizes[1];
  const int INF = 256, HID = 128, NCLS = 16;

  // workspace: norms (2N f32) | H (N*128 f32) | A (N*128 f32) | packed weights (bf16)
  const long fcount = 2L * Nn + 2L * Nn * HID;
  size_t need = (size_t)fcount * 4 + (size_t)(INF * HID + HID * HID + HID * NCLS) * 2;
  if (ws_size < need) return;

  float* ws   = (float*)d_ws;
  float* nrmS = ws;
  float* nrmD = nrmS + Nn;
  float* Hb   = nrmD + Nn;
  float* Ab   = Hb + (long)Nn * HID;
  __bf16* W1p = (__bf16*)(ws + fcount);            // 32B-aligned
  __bf16* W2p = W1p + INF * HID;
  __bf16* W3p = W2p + HID * HID;

  // pack weights into WMMA-B fragment order (one-time, tiny)
  pack_w_kernel<<<(INF * HID + 255) / 256, 256, 0, stream>>>(W1, W1p, INF, HID);
  pack_w_kernel<<<(HID * HID + 255) / 256, 256, 0, stream>>>(W2, W2p, HID, HID);
  pack_w_kernel<<<(HID * NCLS + 255) / 256, 256, 0, stream>>>(W3, W3p, HID, NCLS);

  // degree norms
  zero_kernel<<<256, 256, 0, stream>>>(nrmS, 2L * Nn);
  deg_kernel<<<(E + 255) / 256, 256, 0, stream>>>(src, dst, nrmS, nrmD, E);
  norm_kernel<<<(2 * Nn + 255) / 256, 256, 0, stream>>>(nrmS, 2 * Nn);

  const int gblocks = (Nn + 127) / 128;

  // ---- layer 1: 256 -> 128, ReLU ----
  gemm_norm_kernel<128><<<gblocks, 256, 0, stream>>>(features, W1p, nrmS, Hb, Nn, INF);
  zero_kernel<<<2048, 256, 0, stream>>>(Ab, (long)Nn * HID);
  agg_kernel<32><<<(unsigned)(((long)E * 32 + 255) / 256), 256, 0, stream>>>(Hb, Ab, src, dst, E);
  finish_kernel<128, true><<<(unsigned)(((long)Nn * HID + 255) / 256), 256, 0, stream>>>(Ab, nrmD, b1, Nn);

  // ---- layer 2: 128 -> 128, ReLU ----
  gemm_norm_kernel<128><<<gblocks, 256, 0, stream>>>(Ab, W2p, nrmS, Hb, Nn, HID);
  zero_kernel<<<2048, 256, 0, stream>>>(Ab, (long)Nn * HID);
  agg_kernel<32><<<(unsigned)(((long)E * 32 + 255) / 256), 256, 0, stream>>>(Hb, Ab, src, dst, E);
  finish_kernel<128, true><<<(unsigned)(((long)Nn * HID + 255) / 256), 256, 0, stream>>>(Ab, nrmD, b2, Nn);

  // ---- layer 3: 128 -> 16, no activation, scatter into d_out ----
  gemm_norm_kernel<16><<<gblocks, 256, 0, stream>>>(Ab, W3p, nrmS, Hb, Nn, HID);
  zero_kernel<<<512, 256, 0, stream>>>(out, (long)Nn * NCLS);
  agg_kernel<4><<<(unsigned)(((long)E * 4 + 255) / 256), 256, 0, stream>>>(Hb, out, src, dst, E);
  finish_kernel<16, false><<<(unsigned)(((long)Nn * NCLS + 255) / 256), 256, 0, stream>>>(out, nrmD, b3, Nn);
}
